// DiffLlamaAttention_89739046682920
// MI455X (gfx1250) — matrix operand order repaired and checked by reading, not compiled, not run on hardware
//
#include <hip/hip_runtime.h>
#include <hip/hip_bf16.h>
#include <math.h>

typedef __attribute__((ext_vector_type(16))) _Float16 v16h;
typedef __attribute__((ext_vector_type(8)))  _Float16 v8h;
typedef __attribute__((ext_vector_type(8)))  float    v8f;
typedef __attribute__((ext_vector_type(4)))  int      v4i;

#define B_SZ   2
#define S_LEN  2048
#define HID    2048
#define NH     32
#define NKV    8
#define HD     64
#define LAMBDA_INIT 0.35550906759096927f

#define KS_PAD 72   // Ksh row stride (halves): 144B -> conflict-free A-frag reads
#define VS_PAD 40   // Vsh row stride (halves):  80B -> conflict-free B-frag reads

#if defined(__has_builtin)
#if __has_builtin(__builtin_amdgcn_global_load_async_to_lds_b128) && \
    __has_builtin(__builtin_amdgcn_s_wait_asynccnt)
#define USE_ASYNC_LDS 1
#endif
#endif

__device__ __forceinline__ v8f wmma16(v16h a, v16h b, v8f c) {
  return __builtin_amdgcn_wmma_f32_16x16x32_f16(false, a, false, b, (short)0, c, false, false);
}

#ifdef USE_ASYNC_LDS
typedef __attribute__((address_space(1))) v4i* as1_v4i_ptr;
typedef __attribute__((address_space(3))) v4i* as3_v4i_ptr;
__device__ __forceinline__ void async_cp16(const _Float16* src, _Float16* dst) {
  __builtin_amdgcn_global_load_async_to_lds_b128(
      (as1_v4i_ptr)src, (as3_v4i_ptr)dst, 0, 0);
}
#endif

// ---------------------------------------------------------------- converts
__global__ void f32_to_f16_kernel(const float* __restrict__ in,
                                  _Float16* __restrict__ out, int n) {
  int i = blockIdx.x * blockDim.x + threadIdx.x;
  if (i < n) out[i] = (_Float16)in[i];
}

// ---------------------------------------------------------------- GEMM C = X * W^T
// 256 threads = 8 waves arranged 2(M) x 4(N); wave tile 32x64; WG tile 64 x 256.
template <bool F32OUT>
__global__ __launch_bounds__(256)
void wmma_gemm_xwT(const _Float16* __restrict__ X, const _Float16* __restrict__ W,
                   float* __restrict__ Cf, _Float16* __restrict__ Ch,
                   int M, int N, int K) {
  const int lane = threadIdx.x & 31;
  const int wid  = threadIdx.x >> 5;
  const int mi = wid & 1, ni = wid >> 1;
  const int m0 = blockIdx.x * 64  + mi * 32;
  const int n0 = blockIdx.y * 256 + ni * 64;
  const int l15 = lane & 15, hi = lane >> 4;

  v8f acc[2][4] = {};
  const _Float16* xrow0 = X + (size_t)(m0 + l15) * K;
  const _Float16* xrow1 = xrow0 + (size_t)16 * K;
  for (int k0 = 0; k0 < K; k0 += 32) {
    union { v16h v; v8h h2[2]; } a0, a1;
    a0.h2[0] = *(const v8h*)(xrow0 + k0 + hi * 8);
    a0.h2[1] = *(const v8h*)(xrow0 + k0 + hi * 8 + 16);
    a1.h2[0] = *(const v8h*)(xrow1 + k0 + hi * 8);
    a1.h2[1] = *(const v8h*)(xrow1 + k0 + hi * 8 + 16);
    __builtin_prefetch(xrow0 + k0 + 128, 0, 1);
#pragma unroll
    for (int t = 0; t < 4; ++t) {
      const _Float16* wrow = W + (size_t)(n0 + t * 16 + l15) * K + k0 + hi * 8;
      union { v16h v; v8h h2[2]; } bm;
      bm.h2[0] = *(const v8h*)(wrow);
      bm.h2[1] = *(const v8h*)(wrow + 16);
      acc[0][t] = wmma16(a0.v, bm.v, acc[0][t]);
      acc[1][t] = wmma16(a1.v, bm.v, acc[1][t]);
    }
  }
#pragma unroll
  for (int mm = 0; mm < 2; ++mm) {
#pragma unroll
    for (int t = 0; t < 4; ++t) {
#pragma unroll
      for (int r = 0; r < 8; ++r) {
        size_t idx = (size_t)(m0 + mm * 16 + hi * 8 + r) * N + n0 + t * 16 + l15;
        if (F32OUT) Cf[idx] = acc[mm][t][r];
        else        Ch[idx] = (_Float16)acc[mm][t][r];
      }
    }
  }
}

// ---------------------------------------------------------------- RoPE (in place, f16)
__global__ void rope_kernel(_Float16* __restrict__ X, const float* __restrict__ cosb,
                            const float* __restrict__ sinb, int nheads, int total,
                            float scale) {
  int idx = blockIdx.x * blockDim.x + threadIdx.x;
  if (idx >= total) return;
  int per_row = nheads * 32;
  int rowi = idx / per_row;
  int rem  = idx - rowi * per_row;
  int hh = rem >> 5;
  int d  = rem & 31;
  _Float16* p = X + (size_t)rowi * (nheads * HD) + hh * HD;
  float c = cosb[rowi * HD + d];
  float s = sinb[rowi * HD + d];
  float x1 = (float)p[d], x2 = (float)p[d + 32];
  p[d]      = (_Float16)((x1 * c - x2 * s) * scale);
  p[d + 32] = (_Float16)((x2 * c + x1 * s) * scale);
}

// ---------------------------------------------------------------- V pre-transpose
// VTb[((b*4+kv)*64 + s_blk) * 4096 + d * 32 + s_in]: each 32-key chunk of V^T is
// one contiguous 8KB block (coalesced staging). d<64 -> head kv, d>=64 -> head kv+4.
__global__ __launch_bounds__(256)
void v_transpose_kernel(const _Float16* __restrict__ Vf, _Float16* __restrict__ VTb) {
  __shared__ _Float16 tile[64][65];
  int bkv = blockIdx.z; int b = bkv >> 2, kv = bkv & 3;
  int s0 = blockIdx.x * 64;
  int d0 = blockIdx.y * 64;                       // 0 or 64
  int colbase = (d0 == 0) ? (kv * 64) : ((kv + 4) * 64);
  {
    int dl = threadIdx.x & 63;
    int sl0 = threadIdx.x >> 6;
#pragma unroll
    for (int j = 0; j < 64; j += 4)
      tile[sl0 + j][dl] = Vf[((size_t)(b * S_LEN) + s0 + sl0 + j) * (NKV * HD) + colbase + dl];
  }
  __syncthreads();
  {
    int sl = threadIdx.x & 63;
    int dl0 = threadIdx.x >> 6;
    int s = s0 + sl;
#pragma unroll
    for (int j = 0; j < 64; j += 4) {
      int d = d0 + dl0 + j;
      VTb[((size_t)bkv * 64 + (s >> 5)) * 4096 + d * 32 + (s & 31)] = tile[sl][dl0 + j];
    }
  }
}

// ---------------------------------------------------------------- lambda scalar
__global__ void lambda_kernel(const float* lq1, const float* lk1,
                              const float* lq2, const float* lk2, float* out) {
  int t = threadIdx.x;  // 32 threads
  float s1 = lq1[t] * lk1[t] + lq1[t + 32] * lk1[t + 32];
  float s2 = lq2[t] * lk2[t] + lq2[t + 32] * lk2[t + 32];
#pragma unroll
  for (int m = 16; m; m >>= 1) { s1 += __shfl_xor(s1, m); s2 += __shfl_xor(s2, m); }
  if (t == 0) *out = __expf(s1) - __expf(s2) + LAMBDA_INIT;
}

// ---------------------------------------------------------------- flash attention
// S^T = K * Q^T : C-layout of S^T (row=key r+hi*8, col=query l15) IS the A-fragment
// layout for P*V -> probabilities stay in registers. K/V chunks staged once per WG
// in double-buffered LDS (async-to-LDS when available) shared by all 8 waves.
template <bool MASK>
__device__ __forceinline__ void attn_chunk(
    int kb, int qbase, int l15, int hi,
    const _Float16* __restrict__ ksh,   // Ksh[buf]: [32][KS_PAD]
    const _Float16* __restrict__ vsh,   // Vsh[buf]: [128][VS_PAD]
    const v16h& bq0, const v16h& bq1,
    v8f* acc, float& m_run, float& l_run) {
  const _Float16* kr0 = ksh + l15 * KS_PAD;
  const _Float16* kr1 = ksh + (16 + l15) * KS_PAD;
  union { v16h v; v8h h2[2]; } a00, a01, a10, a11;
  a00.h2[0] = *(const v8h*)(kr0 + hi * 8);
  a00.h2[1] = *(const v8h*)(kr0 + hi * 8 + 16);
  a01.h2[0] = *(const v8h*)(kr0 + 32 + hi * 8);
  a01.h2[1] = *(const v8h*)(kr0 + 32 + hi * 8 + 16);
  a10.h2[0] = *(const v8h*)(kr1 + hi * 8);
  a10.h2[1] = *(const v8h*)(kr1 + hi * 8 + 16);
  a11.h2[0] = *(const v8h*)(kr1 + 32 + hi * 8);
  a11.h2[1] = *(const v8h*)(kr1 + 32 + hi * 8 + 16);

  v8f st0 = {}, st1 = {};
  st0 = wmma16(a00.v, bq0, st0);
  st0 = wmma16(a01.v, bq1, st0);
  st1 = wmma16(a10.v, bq0, st1);
  st1 = wmma16(a11.v, bq1, st1);

  const int query = qbase + l15;
  float s0a[8], s1a[8];
#pragma unroll
  for (int r = 0; r < 8; ++r) {
    s0a[r] = st0[r];
    s1a[r] = st1[r];
    if (MASK) {
      if (kb + hi * 8 + r > query)      s0a[r] = -1e30f;
      if (kb + 16 + hi * 8 + r > query) s1a[r] = -1e30f;
    }
  }
  float mx = -1e30f;
#pragma unroll
  for (int r = 0; r < 8; ++r) mx = fmaxf(mx, fmaxf(s0a[r], s1a[r]));
  mx = fmaxf(mx, __shfl_xor(mx, 16));
  float mn = fmaxf(m_run, mx);
  float alpha = __expf(m_run - mn);
  float rs = 0.f;
  union { v16h v; _Float16 e[16]; } ap;
#pragma unroll
  for (int r = 0; r < 8; ++r) {
    float p0 = __expf(s0a[r] - mn);
    float p1 = __expf(s1a[r] - mn);
    rs += p0 + p1;
    ap.e[r]     = (_Float16)p0;   // A elements 0..7  <-> keys kb + hi*8 + r
    ap.e[8 + r] = (_Float16)p1;   // A elements 8..15 <-> keys kb + 16 + hi*8 + r
  }
  rs += __shfl_xor(rs, 16);
  l_run = l_run * alpha + rs;
  m_run = mn;
#pragma unroll
  for (int r = 0; r < 8; ++r) {
    float al = __shfl(alpha, hi * 8 + r);
#pragma unroll
    for (int t = 0; t < 8; ++t) acc[t][r] = acc[t][r] * al;
  }
#pragma unroll
  for (int t = 0; t < 8; ++t) {
    const _Float16* vp = vsh + (t * 16 + l15) * VS_PAD + hi * 8;
    union { v16h v; v8h h2[2]; } bv;
    bv.h2[0] = *(const v8h*)(vp);
    bv.h2[1] = *(const v8h*)(vp + 16);
    acc[t] = wmma16(ap.v, bv.v, acc[t]);
  }
}

// grid: (S/128, NH, B); 256 threads = 8 waves; each wave owns a 16-row Q tile.
__global__ __launch_bounds__(256)
void flash_attn_kernel(const _Float16* __restrict__ Qf, const _Float16* __restrict__ Kf,
                       const _Float16* __restrict__ VTb, _Float16* __restrict__ AO) {
  const int tid  = threadIdx.x;
  const int lane = tid & 31;
  const int b = blockIdx.z, h = blockIdx.y;
  const int qwg = blockIdx.x * 128;
  // scalarize the per-wave tile id so the mask branch is a true scalar branch
  const int wid_u  = __builtin_amdgcn_readfirstlane(tid >> 5);
  const int qbase  = qwg + wid_u * 16;
  const int l15 = lane & 15, hi = lane >> 4;
  const int kvQK = h >> 2;           // K head (GQA)
  const int pair = (h & 15) >> 2;    // concatenated-V pair index

  __shared__ __align__(16) _Float16 Ksh[2][32 * KS_PAD];
  __shared__ __align__(16) _Float16 Vsh[2][128 * VS_PAD];

  const _Float16* kglob = Kf + ((size_t)(b * S_LEN)) * (NKV * HD) + kvQK * HD;
  const _Float16* vglob = VTb + ((size_t)(b * 4 + pair) * 64) * 4096;

  // cooperative staging: K chunk 32x64 (4KB), V^T chunk 128x32 (8KB)
  auto stage = [&](int buf, int kb) {
    int row = tid >> 3, seg = (tid & 7) * 8;     // 256 threads: one 16B each (K)
    const _Float16* ksrc = kglob + (size_t)(kb + row) * (NKV * HD) + seg;
    _Float16* kdst = &Ksh[buf][row * KS_PAD + seg];
    int d = tid >> 1, o = (tid & 1) * 16;        // 256 threads: 32B each (V)
    const _Float16* vsrc = vglob + (size_t)(kb >> 5) * 4096 + d * 32 + o;
    _Float16* vdst = &Vsh[buf][d * VS_PAD + o];
#ifdef USE_ASYNC_LDS
    async_cp16(ksrc, kdst);
    async_cp16(vsrc, vdst);
    async_cp16(vsrc + 8, vdst + 8);
#else
    *(v8h*)kdst       = *(const v8h*)ksrc;
    *(v8h*)vdst       = *(const v8h*)vsrc;
    *(v8h*)(vdst + 8) = *(const v8h*)(vsrc + 8);
#endif
  };
  auto fence = [&]() {
#ifdef USE_ASYNC_LDS
    __builtin_amdgcn_s_wait_asynccnt(0);
#endif
    __syncthreads();
  };

  // Q B-fragments (loop invariant): B[k=d][n=query]
  const _Float16* qrow = Qf + ((size_t)(b * S_LEN) + qbase + l15) * (NH * HD) + h * HD;
  union { v16h v; v8h h2[2]; } bq0, bq1;
  bq0.h2[0] = *(const v8h*)(qrow + hi * 8);
  bq0.h2[1] = *(const v8h*)(qrow + hi * 8 + 16);
  bq1.h2[0] = *(const v8h*)(qrow + 32 + hi * 8);
  bq1.h2[1] = *(const v8h*)(qrow + 32 + hi * 8 + 16);

  v8f acc[8] = {};
  float m_run = -1e30f, l_run = 0.f;

  const int kend = qwg + 128;      // WG-uniform trip count (barrier-safe)
  stage(0, 0);
  fence();
  int buf = 0;
  for (int kb = 0; kb < kend; kb += 32) {
    if (kb + 32 < kend) stage(buf ^ 1, kb + 32);
    if (kb + 31 <= qbase)          // scalar (wave-uniform) branch, no barriers inside
      attn_chunk<false>(kb, qbase, l15, hi, Ksh[buf], Vsh[buf], bq0.v, bq1.v,
                        acc, m_run, l_run);
    else
      attn_chunk<true>(kb, qbase, l15, hi, Ksh[buf], Vsh[buf], bq0.v, bq1.v,
                       acc, m_run, l_run);
    fence();
    buf ^= 1;
  }

  // epilogue: normalize by row sum, store AO (b,h,s,128)
  float rinv = 1.0f / l_run;
  _Float16* ao = AO + ((size_t)(b * NH + h)) * S_LEN * 128;
#pragma unroll
  for (int r = 0; r < 8; ++r) {
    float rl = __shfl(rinv, hi * 8 + r);
    int qr = qbase + r + hi * 8;
#pragma unroll
    for (int t = 0; t < 8; ++t)
      ao[(size_t)qr * 128 + t * 16 + l15] = (_Float16)(acc[t][r] * rl);
  }
}

// ---------------------------------------------------------------- diff + RMS norm
__global__ __launch_bounds__(128)
void diff_norm_kernel(const _Float16* __restrict__ AO, const float* __restrict__ lam_p,
                      _Float16* __restrict__ OUTN) {
  int bid = blockIdx.x;
  int hp = bid & 15;
  int bs = bid >> 4;
  int b = bs >> 11, s = bs & (S_LEN - 1);
  int d = threadIdx.x;
  float lam = *lam_p;
  size_t i1 = ((size_t)(b * NH + hp)      * S_LEN + s) * 128 + d;
  size_t i2 = ((size_t)(b * NH + hp + 16) * S_LEN + s) * 128 + d;
  float o = (float)AO[i1] - lam * (float)AO[i2];
  float ss = o * o;
#pragma unroll
  for (int m = 16; m; m >>= 1) ss += __shfl_xor(ss, m);
  __shared__ float red[4];
  if ((threadIdx.x & 31) == 0) red[threadIdx.x >> 5] = ss;
  __syncthreads();
  float tot = red[0] + red[1] + red[2] + red[3];
  float r = rsqrtf(tot * (1.0f / 128.0f) + 1e-6f);
  OUTN[(size_t)bs * 2048 + hp * 128 + d] = (_Float16)((1.0f - LAMBDA_INIT) * o * r);
}

// ---------------------------------------------------------------- launch
extern "C" void kernel_launch(void* const* d_in, const int* in_sizes, int n_in,
                              void* d_out, int out_size, void* d_ws, size_t ws_size,
                              hipStream_t stream) {
  const float* hidden = (const float*)d_in[0];
  const float* cosb   = (const float*)d_in[1];
  const float* sinb   = (const float*)d_in[2];
  // d_in[3] attention_mask: causal, implemented analytically
  const float* wq  = (const float*)d_in[4];
  const float* wk  = (const float*)d_in[5];
  const float* wv  = (const float*)d_in[6];
  const float* wo  = (const float*)d_in[7];
  const float* lq1 = (const float*)d_in[8];
  const float* lk1 = (const float*)d_in[9];
  const float* lq2 = (const float*)d_in[10];
  const float* lk2 = (const float*)d_in[11];
  float* out = (float*)d_out;

  const int M = B_SZ * S_LEN;          // 4096
  const size_t nXh = (size_t)M * HID;  // 8,388,608
  const size_t nWq = (size_t)HID * HID;
  const size_t nWk = (size_t)(NKV * HD) * HID;
  const size_t nKV = (size_t)M * NKV * HD;
  const size_t nVT = (size_t)B_SZ * 4 * 128 * S_LEN;
  const size_t nAO = (size_t)B_SZ * NH * S_LEN * 128;

  char* ws = (char*)d_ws;
  size_t off = 0;
  _Float16* Xh = (_Float16*)(ws + off); off += nXh * 2;
  _Float16* Wq = (_Float16*)(ws + off); off += nWq * 2;
  _Float16* Wk = (_Float16*)(ws + off); off += nWk * 2;
  _Float16* Wv = (_Float16*)(ws + off); off += nWk * 2;
  _Float16* Wo = (_Float16*)(ws + off); off += nWq * 2;
  _Float16* Qf = (_Float16*)(ws + off); off += nXh * 2;
  _Float16* Kf = (_Float16*)(ws + off); off += nKV * 2;
  _Float16* Vf = (_Float16*)(ws + off); off += nKV * 2;
  _Float16* VT = (_Float16*)(ws + off); off += nVT * 2;
  _Float16* AO = (_Float16*)(ws + off); off += nAO * 2;
  _Float16* ON = (_Float16*)(ws + off); off += nXh * 2;
  float* lam   = (float*)(ws + off);

  // 1) f32 -> f16 conversions
  f32_to_f16_kernel<<<(nXh + 255) / 256, 256, 0, stream>>>(hidden, Xh, (int)nXh);
  f32_to_f16_kernel<<<(nWq + 255) / 256, 256, 0, stream>>>(wq, Wq, (int)nWq);
  f32_to_f16_kernel<<<(nWk + 255) / 256, 256, 0, stream>>>(wk, Wk, (int)nWk);
  f32_to_f16_kernel<<<(nWk + 255) / 256, 256, 0, stream>>>(wv, Wv, (int)nWk);
  f32_to_f16_kernel<<<(nWq + 255) / 256, 256, 0, stream>>>(wo, Wo, (int)nWq);

  // 2) projections (WMMA)
  wmma_gemm_xwT<false><<<dim3(M / 64, HID / 256), 256, 0, stream>>>(
      Xh, Wq, nullptr, Qf, M, HID, HID);
  wmma_gemm_xwT<false><<<dim3(M / 64, (NKV * HD) / 256), 256, 0, stream>>>(
      Xh, Wk, nullptr, Kf, M, NKV * HD, HID);
  wmma_gemm_xwT<false><<<dim3(M / 64, (NKV * HD) / 256), 256, 0, stream>>>(
      Xh, Wv, nullptr, Vf, M, NKV * HD, HID);

  // 3) RoPE (Q pre-scaled by 1/sqrt(HD))
  int nq = M * NH * 32, nk = M * NKV * 32;
  rope_kernel<<<(nq + 255) / 256, 256, 0, stream>>>(Qf, cosb, sinb, NH, nq, 0.125f);
  rope_kernel<<<(nk + 255) / 256, 256, 0, stream>>>(Kf, cosb, sinb, NKV, nk, 1.0f);

  // 4) V transpose (blocked layout) + lambda
  v_transpose_kernel<<<dim3(S_LEN / 64, 2, B_SZ * 4), 256, 0, stream>>>(Vf, VT);
  lambda_kernel<<<1, 32, 0, stream>>>(lq1, lk1, lq2, lk2, lam);

  // 5) attention (WMMA flash, async-LDS-shared K/V, register-resident P)
  flash_attn_kernel<<<dim3(S_LEN / 128, NH, B_SZ), 256, 0, stream>>>(Qf, Kf, VT, AO);

  // 6) differential combine + RMS norm
  diff_norm_kernel<<<M * 16, 128, 0, stream>>>(AO, lam, ON);

  // 7) output projection (WMMA, f32 out)
  wmma_gemm_xwT<true><<<dim3(M / 64, HID / 256), 256, 0, stream>>>(
      ON, Wo, out, nullptr, M, HID, HID);
}
